// MultiHeadAttention_31155692765522
// MI455X (gfx1250) — matrix operand order repaired and checked by reading, not compiled, not run on hardware
//
#include <hip/hip_runtime.h>

// ---------------------------------------------------------------------------
// MI455X (gfx1250) fused multi-head attention: bf16 WMMA with fp32 accumulate,
// async global->LDS tile staging (ASYNCcnt) and double-buffered LDS tiles.
// ---------------------------------------------------------------------------

typedef __attribute__((ext_vector_type(16))) __bf16 v16bf;
typedef __attribute__((ext_vector_type(8)))  __bf16 v8bf;
typedef __attribute__((ext_vector_type(8)))  float  v8f;
typedef int v4i_vs __attribute__((vector_size(16)));  // matches builtin param

#define WMMA_BF16(a, b, c) \
  __builtin_amdgcn_wmma_f32_16x16x32_bf16(false, (a), false, (b), (short)0, (c), false, false)

#if __has_builtin(__builtin_amdgcn_global_load_async_to_lds_b128)
#define HAS_ASYNC_LDS 1
#else
#define HAS_ASYNC_LDS 0
#endif

// --- small device helpers ---------------------------------------------------

// Async copy of 16 bytes global->LDS (no VGPR round trip; tracked by ASYNCcnt).
__device__ __forceinline__ void async_cp16B(const void* g, void* l) {
#if HAS_ASYNC_LDS
  __builtin_amdgcn_global_load_async_to_lds_b128(
      (__attribute__((address_space(1))) v4i_vs*)(unsigned long long)g,
      (__attribute__((address_space(3))) v4i_vs*)(unsigned)(unsigned long long)l,
      0, 0);
#else
  *(v8bf*)l = *(const v8bf*)g;
#endif
}

__device__ __forceinline__ void wait_async_zero() {
#if __has_builtin(__builtin_amdgcn_s_wait_asynccnt)
  __builtin_amdgcn_s_wait_asynccnt(0);
#elif HAS_ASYNC_LDS
  asm volatile("s_wait_asynccnt 0" ::: "memory");
#endif
}

// Convert 16 contiguous fp32 -> 16 bf16 at dst (LDS staging).
__device__ __forceinline__ void cvt_store16(const float* __restrict__ src,
                                            __bf16* __restrict__ dst) {
#pragma unroll
  for (int i = 0; i < 16; i += 4) {
    float4 v = *(const float4*)(src + i);
    dst[i + 0] = (__bf16)v.x;
    dst[i + 1] = (__bf16)v.y;
    dst[i + 2] = (__bf16)v.z;
    dst[i + 3] = (__bf16)v.w;
  }
}

// Copy 16 contiguous bf16 (two 16B vectors) through VGPRs.
__device__ __forceinline__ void copy16(const __bf16* __restrict__ src,
                                       __bf16* __restrict__ dst) {
  *(v8bf*)dst       = *(const v8bf*)src;
  *(v8bf*)(dst + 8) = *(const v8bf*)(src + 8);
}

// Stage 16 bf16 global->LDS, async when available.
__device__ __forceinline__ void stage16(const __bf16* __restrict__ src,
                                        __bf16* __restrict__ dst) {
#if HAS_ASYNC_LDS
  async_cp16B(src, dst);
  async_cp16B(src + 8, dst + 8);
#else
  copy16(src, dst);
#endif
}

// A-matrix fragment (16x32 bf16): lane holds K = off..off+15
// of its row, where off = (lane>>4)*16.  p points at (row, off).
__device__ __forceinline__ v16bf ld_fragA(const __bf16* __restrict__ p) {
  v16bf f;
#pragma unroll
  for (int i = 0; i < 8; ++i) { f[i] = p[i]; f[i + 8] = p[8 + i]; }
  return f;
}

// B-matrix fragment (32x16 bf16): lane holds 16 contiguous K values of its
// column, starting at K = (lane>>4)*16.  p points at (col, koff).
__device__ __forceinline__ v16bf ld_fragB(const __bf16* __restrict__ p) {
  v16bf f;
#pragma unroll
  for (int i = 0; i < 16; ++i) f[i] = p[i];
  return f;
}

// ---------------------------------------------------------------------------
// Generic NT GEMM:  C[M,N] = A[M,K] * B[N,K]^T + bias[N]   (torch Linear)
//   A_BF16: A operand is bf16 (staged async), else fp32 (converted in staging)
//   EPI: 0 = bf16 out [M,N], 1 = bf16 out transposed [N,M], 2 = fp32 out [M,N]
// Block: 256 threads = 8 waves; block tile 128x128, k-tile 32, double-buffered
// LDS (one barrier per k-step).  Wave (wm,wn) owns a 32x64 tile = 2x4 WMMAs.
// ---------------------------------------------------------------------------
template <bool A_BF16, int EPI>
__global__ __launch_bounds__(256) void gemm_nt_kernel(
    const void* __restrict__ Ap, const float* __restrict__ Bw,
    const float* __restrict__ bias, void* __restrict__ Cp,
    int M, int N, int K) {
  constexpr int LDT = 40;  // padded bf16 row stride (80B) -> conflict-free
  __shared__ __bf16 As[2][128 * LDT];
  __shared__ __bf16 Bs[2][128 * LDT];

  const int tid  = threadIdx.x;
  const int lane = tid & 31;
  const int wid  = tid >> 5;
  const int wm   = wid & 3;   // wave row block (32 rows)
  const int wn   = wid >> 2;  // wave col block (64 cols)
  const int m0   = blockIdx.x * 128;
  const int n0   = blockIdx.y * 128;

  const int mm  = lane & 15;
  const int koA = (lane >> 4) << 4;  // A fragment K offset (0 / 16)
  const int koB = (lane >> 4) << 4;  // B fragment K offset (0 / 16)

  v8f acc[2][4] = {};

  const int lrow  = tid >> 1;        // 0..127 : tile row staged by this thread
  const int lhalf = (tid & 1) * 16;  // 0/16   : K half staged by this thread

  auto stage = [&](int k0, int pb) {
    if (A_BF16) {
      const __bf16* A = (const __bf16*)Ap;
      stage16(A + (size_t)(m0 + lrow) * K + k0 + lhalf,
              &As[pb][lrow * LDT + lhalf]);
    } else {
      const float* A = (const float*)Ap;
      cvt_store16(A + (size_t)(m0 + lrow) * K + k0 + lhalf,
                  &As[pb][lrow * LDT + lhalf]);
    }
    // B (weights) is fp32: conversion must go through VGPRs.
    cvt_store16(Bw + (size_t)(n0 + lrow) * K + k0 + lhalf,
                &Bs[pb][lrow * LDT + lhalf]);
  };

  stage(0, 0);
  int pb = 0;
  for (int k0 = 0; k0 < K; k0 += 32) {
    if (A_BF16) wait_async_zero();  // this wave's async LDS writes landed
    __syncthreads();                // everyone's staging of buffer pb done
    if (k0 + 32 < K) stage(k0 + 32, pb ^ 1);  // overlap with compute

    v16bf af[2], bf[4];
#pragma unroll
    for (int mt = 0; mt < 2; ++mt)
      af[mt] = ld_fragA(&As[pb][(wm * 32 + mt * 16 + mm) * LDT + koA]);
#pragma unroll
    for (int nt = 0; nt < 4; ++nt)
      bf[nt] = ld_fragB(&Bs[pb][(wn * 64 + nt * 16 + mm) * LDT + koB]);

#pragma unroll
    for (int mt = 0; mt < 2; ++mt)
#pragma unroll
      for (int nt = 0; nt < 4; ++nt)
        acc[mt][nt] = WMMA_BF16(af[mt], bf[nt], acc[mt][nt]);
    pb ^= 1;
  }

  // ---- epilogue: C layout = VGPR i -> row (i | i+8 by half), col = lane&15 --
  const int rhalf = (lane >> 4) << 3;
  float bv[4];
#pragma unroll
  for (int nt = 0; nt < 4; ++nt)
    bv[nt] = bias[n0 + wn * 64 + nt * 16 + mm];

  if (EPI == 2) {
    float* C = (float*)Cp;
#pragma unroll
    for (int mt = 0; mt < 2; ++mt)
#pragma unroll
      for (int nt = 0; nt < 4; ++nt) {
        const int c = n0 + wn * 64 + nt * 16 + mm;
#pragma unroll
        for (int i = 0; i < 8; ++i) {
          const int r = m0 + wm * 32 + mt * 16 + rhalf + i;
          C[(size_t)r * N + c] = acc[mt][nt][i] + bv[nt];
        }
      }
  } else if (EPI == 0) {
    __bf16* C = (__bf16*)Cp;
#pragma unroll
    for (int mt = 0; mt < 2; ++mt)
#pragma unroll
      for (int nt = 0; nt < 4; ++nt) {
        const int c = n0 + wn * 64 + nt * 16 + mm;
#pragma unroll
        for (int i = 0; i < 8; ++i) {
          const int r = m0 + wm * 32 + mt * 16 + rhalf + i;
          C[(size_t)r * N + c] = (__bf16)(acc[mt][nt][i] + bv[nt]);
        }
      }
  } else {  // EPI == 1 : transposed bf16, Ct[c*M + r]; 8 consecutive rows/lane
    __bf16* Ct = (__bf16*)Cp;
#pragma unroll
    for (int mt = 0; mt < 2; ++mt)
#pragma unroll
      for (int nt = 0; nt < 4; ++nt) {
        const int c  = n0 + wn * 64 + nt * 16 + mm;
        const int r8 = m0 + wm * 32 + mt * 16 + rhalf;
        v8bf pk;
#pragma unroll
        for (int i = 0; i < 8; ++i) pk[i] = (__bf16)(acc[mt][nt][i] + bv[nt]);
        *(v8bf*)&Ct[(size_t)c * M + r8] = pk;  // 16B packed store
      }
  }
}

// ---------------------------------------------------------------------------
// Flash attention over bf16 Q/K (row-major [4096,768]) and V^T ([768,4096]).
// Grid: (S/64, HEADS, B). Block: 128 threads = 4 waves; wave owns 16 q rows.
// kv tiles of 64, double-buffered in LDS and staged with async global->LDS;
// online softmax in WMMA C-register layout.
// ---------------------------------------------------------------------------
__global__ __launch_bounds__(128) void attn_kernel(
    const void* __restrict__ Qp, const void* __restrict__ Kp,
    const void* __restrict__ Vtp, void* __restrict__ Op) {
  constexpr int S = 2048, E = 768, D = 64;
  constexpr int LKV = 72;  // padded bf16 row stride (144B) -> conflict-free
  constexpr int LP  = 40;
  __shared__ __bf16 Ks[2][64 * LKV];   // [kv][d]
  __shared__ __bf16 Vs[2][64 * LKV];   // [d][kv]   (from V^T)
  __shared__ __bf16 Ps[4][16 * LP];    // per-wave P staging [16 q][64 kv]

  const __bf16* Q  = (const __bf16*)Qp;
  const __bf16* Kb = (const __bf16*)Kp;
  const __bf16* Vt = (const __bf16*)Vtp;
  __bf16*       O  = (__bf16*)Op;

  const int tid  = threadIdx.x;
  const int lane = tid & 31;
  const int wid  = tid >> 5;
  const int h = blockIdx.y, b = blockIdx.z;
  const int q0 = blockIdx.x * 64 + wid * 16;

  const int mm    = lane & 15;
  const int koA   = (lane >> 4) << 4;
  const int koB   = (lane >> 4) << 4;
  const int rhalf = (lane >> 4) << 3;

  // Q fragments for this wave's 16 rows (K dim = d, two 32-wide k-steps).
  v16bf qf[2];
  {
    const __bf16* qrow = Q + (size_t)(b * S + q0 + mm) * E + h * D;
#pragma unroll
    for (int ks = 0; ks < 2; ++ks) qf[ks] = ld_fragA(qrow + ks * 32 + koA);
  }

  v8f  o[4] = {};
  float mrun[8], lsum[8];
#pragma unroll
  for (int i = 0; i < 8; ++i) { mrun[i] = -3.0e38f; lsum[i] = 0.0f; }

  const float sl2e = 0.125f * 1.44269504088896340736f;  // 1/sqrt(64) * log2(e)

  const int crow  = tid >> 1;        // 0..63 : tile row staged by this thread
  const int chalf = (tid & 1) * 32;  // 0/32

  auto stage_kv = [&](int t, int pb) {
    const __bf16* ksrc = Kb + (size_t)(b * S + t + crow) * E + h * D + chalf;
    __bf16*       kdst = &Ks[pb][crow * LKV + chalf];
    stage16(ksrc,      kdst);
    stage16(ksrc + 16, kdst + 16);
    const __bf16* vsrc = Vt + (size_t)(h * D + crow) * (2 * S) + b * S + t + chalf;
    __bf16*       vdst = &Vs[pb][crow * LKV + chalf];
    stage16(vsrc,      vdst);
    stage16(vsrc + 16, vdst + 16);
  };

  stage_kv(0, 0);
  int pb = 0;
  for (int t = 0; t < S; t += 64) {
    wait_async_zero();  // this wave's async LDS writes landed
    __syncthreads();    // all waves staged buffer pb; prior reads of pb^1 done
    if (t + 64 < S) stage_kv(t + 64, pb ^ 1);  // overlap with compute

    // ---- scores S = Q * K^T  (B column n = kv row n, contiguous in d) ----
    v8f s[4] = {};
#pragma unroll
    for (int nt = 0; nt < 4; ++nt)
#pragma unroll
      for (int ks = 0; ks < 2; ++ks) {
        v16bf kf = ld_fragB(&Ks[pb][(nt * 16 + mm) * LKV + ks * 32 + koB]);
        s[nt] = WMMA_BF16(qf[ks], kf, s[nt]);
      }

    // ---- online softmax (per C-register row i; halves are independent) ----
#pragma unroll
    for (int nt = 0; nt < 4; ++nt) s[nt] = s[nt] * sl2e;
#pragma unroll
    for (int i = 0; i < 8; ++i) {
      float mx = fmaxf(fmaxf(s[0][i], s[1][i]), fmaxf(s[2][i], s[3][i]));
      mx = fmaxf(mx, __shfl_xor(mx, 1));
      mx = fmaxf(mx, __shfl_xor(mx, 2));
      mx = fmaxf(mx, __shfl_xor(mx, 4));
      mx = fmaxf(mx, __shfl_xor(mx, 8));
      const float mn   = fmaxf(mrun[i], mx);
      const float corr = exp2f(mrun[i] - mn);
      mrun[i] = mn;
      float rs = 0.0f;
#pragma unroll
      for (int nt = 0; nt < 4; ++nt) {
        const float p = exp2f(s[nt][i] - mn);
        s[nt][i] = p;
        rs += p;
      }
      rs += __shfl_xor(rs, 1);
      rs += __shfl_xor(rs, 2);
      rs += __shfl_xor(rs, 4);
      rs += __shfl_xor(rs, 8);
      lsum[i] = lsum[i] * corr + rs;
#pragma unroll
      for (int dt = 0; dt < 4; ++dt) o[dt][i] = o[dt][i] * corr;
      // stage P (bf16) into this wave's private LDS region
      const int pr = i + rhalf;
#pragma unroll
      for (int nt = 0; nt < 4; ++nt)
        Ps[wid][pr * LP + nt * 16 + mm] = (__bf16)s[nt][i];
    }
    // wave-private LDS: DS ops from one wave are in-order; drain the counter.
    asm volatile("s_wait_dscnt 0" ::: "memory");

    // ---- O += P * V  (B column n = d row of V^T, contiguous in kv) ----
    v16bf pf[2];
#pragma unroll
    for (int ks = 0; ks < 2; ++ks)
      pf[ks] = ld_fragA(&Ps[wid][mm * LP + ks * 32 + koA]);
#pragma unroll
    for (int dt = 0; dt < 4; ++dt)
#pragma unroll
      for (int ks = 0; ks < 2; ++ks) {
        v16bf vf = ld_fragB(&Vs[pb][(dt * 16 + mm) * LKV + ks * 32 + koB]);
        o[dt] = WMMA_BF16(pf[ks], vf, o[dt]);
      }
    pb ^= 1;
  }

  // ---- epilogue: divide by l, write bf16 attn output [b,s,h*64+d] ----
#pragma unroll
  for (int i = 0; i < 8; ++i) {
    const float inv = 1.0f / lsum[i];
    const int   r   = q0 + i + rhalf;
#pragma unroll
    for (int dt = 0; dt < 4; ++dt) {
      const int c = h * D + dt * 16 + mm;
      O[(size_t)(b * S + r) * E + c] = (__bf16)(o[dt][i] * inv);
    }
  }
}

// ---------------------------------------------------------------------------
// Host launch
// ---------------------------------------------------------------------------
extern "C" void kernel_launch(void* const* d_in, const int* in_sizes, int n_in,
                              void* d_out, int out_size, void* d_ws, size_t ws_size,
                              hipStream_t stream) {
  constexpr int B = 2, S = 2048, E = 768;
  constexpr int M = B * S;  // 4096
  const float* x  = (const float*)d_in[0];
  const float* wq = (const float*)d_in[1];
  const float* bq = (const float*)d_in[2];
  const float* wk = (const float*)d_in[3];
  const float* bk = (const float*)d_in[4];
  const float* wv = (const float*)d_in[5];
  const float* bv = (const float*)d_in[6];
  const float* wo = (const float*)d_in[7];
  const float* bo = (const float*)d_in[8];
  float* out = (float*)d_out;

  // workspace: 4 bf16 buffers of 4096*768 (Q, K, V^T, attn-out) = 25.2 MB
  const size_t NELT = (size_t)M * E;
  unsigned short* ws = (unsigned short*)d_ws;
  void* qb  = (void*)(ws);
  void* kb  = (void*)(ws + NELT);
  void* vtb = (void*)(ws + 2 * NELT);
  void* ab  = (void*)(ws + 3 * NELT);

  dim3 gg(M / 128, E / 128);  // (32, 6)
  dim3 gb(256);
  gemm_nt_kernel<false, 0><<<gg, gb, 0, stream>>>(x, wq, bq, qb,  M, E, E);
  gemm_nt_kernel<false, 0><<<gg, gb, 0, stream>>>(x, wk, bk, kb,  M, E, E);
  gemm_nt_kernel<false, 1><<<gg, gb, 0, stream>>>(x, wv, bv, vtb, M, E, E);

  dim3 ga(S / 64, 12, B);  // (32, 12, 2)
  attn_kernel<<<ga, dim3(128), 0, stream>>>(qb, kb, vtb, ab);

  gemm_nt_kernel<true, 2><<<gg, gb, 0, stream>>>(ab, wo, bo, out, M, E, E);
}
